// GraphNetwork_25091198943371
// MI455X (gfx1250) — matrix operand order repaired and checked by reading, not compiled, not run on hardware
//
#include <hip/hip_runtime.h>
#include <hip/hip_bf16.h>

#define N_NODES 8192
#define N_FEAT  512
#define HID     64
#define N_REL   16
#define N_BASES 30
#define B_CONV  64
#define SEQ_L   128
#define N_CLS   7
#define N_EDGE  524288
#define D_EMO   576
#define WCAT_N  1088   /* 16 relations * 64 + root slot (64) */

typedef __bf16 bf16;
typedef __attribute__((ext_vector_type(16))) __bf16 v16bf;
typedef __attribute__((ext_vector_type(8)))  __bf16 v8bf;
typedef __attribute__((ext_vector_type(8)))  float  v8f;

// Fragment loaders matching the CDNA5 16-bit A(16x32)/B(32x16) VGPR layouts.
// A: lane holds row m; K slices [kk+hi*8, +8) and [kk+16+hi*8, +8).
__device__ __forceinline__ v16bf load_afrag(const bf16* __restrict__ arow, int kk, int hi)
{
    const v8bf a0 = *(const v8bf*)(arow + kk + hi * 8);
    const v8bf a1 = *(const v8bf*)(arow + kk + 16 + hi * 8);
    v16bf a;
#pragma unroll
    for (int e = 0; e < 8; ++e) { a[e] = a0[e]; a[8 + e] = a1[e]; }
    return a;
}
// B: lane holds col n; K slices [kk+hi*8, +8) and [kk+16+hi*8, +8).
__device__ __forceinline__ v16bf load_bfrag(const bf16* __restrict__ brow, int kk, int hi)
{
    const v8bf b0 = *(const v8bf*)(brow + kk + hi * 8);
    const v8bf b1 = *(const v8bf*)(brow + kk + 16 + hi * 8);
    v16bf b;
#pragma unroll
    for (int e = 0; e < 8; ++e) { b[e] = b0[e]; b[8 + e] = b1[e]; }
    return b;
}

// ---------------------------------------------------------------------------
// Generic bf16 WMMA GEMM:  C[M,N] = A[M,K] @ Bt[N,K]^T   (+bias, opt relu)
// A, Bt row-major bf16, lda=ldb=K, ldc=N.  Batched via blockIdx.y.
// One wave computes a 16x64 C tile (4 accumulators); the K loop is software-
// pipelined with explicit double buffering so the next step's 10 b128 loads
// are in flight while the current 4 v_wmma execute (partial s_wait_loadcnt
// instead of a full drain before every wmma).
// Requires: M % 16 == 0, N % 64 == 0, K % 32 == 0 (true for all call sites).
// ---------------------------------------------------------------------------
__global__ void gemm_bf16_wmma(const bf16* __restrict__ A,
                               const bf16* __restrict__ Bt,
                               float* __restrict__ Cf,
                               bf16*  __restrict__ Cb,
                               const float* __restrict__ bias,
                               int M, int N, int K,
                               long long aStride, long long bStride, long long cStride,
                               int relu)
{
    const int lane = threadIdx.x & 31;
    const int hi   = lane >> 4;        // which half of the wave
    const int nl   = lane & 15;
    const int tilesN = N >> 6;         // 64-wide N tiles
    const int tiles  = (M >> 4) * tilesN;
    const int tile = blockIdx.x * (blockDim.x >> 5) + (threadIdx.x >> 5);
    if (tile >= tiles) return;         // wave-uniform: EXEC stays all-ones
    const int ti = tile / tilesN;
    const int tj = tile % tilesN;
    const long long batch = blockIdx.y;

    const bf16* __restrict__ arow = A  + batch * aStride + (long long)(ti * 16 + nl) * K;
    const bf16* __restrict__ br0  = Bt + batch * bStride + (long long)(tj * 64 + nl) * K;
    const bf16* __restrict__ br1  = br0 + (long long)16 * K;
    const bf16* __restrict__ br2  = br0 + (long long)32 * K;
    const bf16* __restrict__ br3  = br0 + (long long)48 * K;

    v8f c[4];
#pragma unroll
    for (int t = 0; t < 4; ++t)
#pragma unroll
        for (int i = 0; i < 8; ++i) c[t][i] = 0.0f;

    // ---- prologue: fragments for kk = 0 ----
    v16bf a  = load_afrag(arow, 0, hi);
    v16bf b0 = load_bfrag(br0, 0, hi);
    v16bf b1 = load_bfrag(br1, 0, hi);
    v16bf b2 = load_bfrag(br2, 0, hi);
    v16bf b3 = load_bfrag(br3, 0, hi);

    // ---- pipelined main loop: prefetch kk, compute kk-32 ----
#pragma unroll 2
    for (int kk = 32; kk < K; kk += 32) {
        v16bf an  = load_afrag(arow, kk, hi);
        v16bf bn0 = load_bfrag(br0, kk, hi);
        v16bf bn1 = load_bfrag(br1, kk, hi);
        v16bf bn2 = load_bfrag(br2, kk, hi);
        v16bf bn3 = load_bfrag(br3, kk, hi);
        c[0] = __builtin_amdgcn_wmma_f32_16x16x32_bf16(false, a, false, b0, (short)0, c[0], false, false);
        c[1] = __builtin_amdgcn_wmma_f32_16x16x32_bf16(false, a, false, b1, (short)0, c[1], false, false);
        c[2] = __builtin_amdgcn_wmma_f32_16x16x32_bf16(false, a, false, b2, (short)0, c[2], false, false);
        c[3] = __builtin_amdgcn_wmma_f32_16x16x32_bf16(false, a, false, b3, (short)0, c[3], false, false);
        a = an; b0 = bn0; b1 = bn1; b2 = bn2; b3 = bn3;
    }
    // ---- epilogue: last K step ----
    c[0] = __builtin_amdgcn_wmma_f32_16x16x32_bf16(false, a, false, b0, (short)0, c[0], false, false);
    c[1] = __builtin_amdgcn_wmma_f32_16x16x32_bf16(false, a, false, b1, (short)0, c[1], false, false);
    c[2] = __builtin_amdgcn_wmma_f32_16x16x32_bf16(false, a, false, b2, (short)0, c[2], false, false);
    c[3] = __builtin_amdgcn_wmma_f32_16x16x32_bf16(false, a, false, b3, (short)0, c[3], false, false);

    float* cf = Cf ? Cf + batch * cStride : nullptr;
    bf16*  cb = Cb ? Cb + batch * cStride : nullptr;
#pragma unroll
    for (int t = 0; t < 4; ++t) {
        const int n = tj * 64 + t * 16 + nl;
#pragma unroll
        for (int j0 = 0; j0 < 8; ++j0) {
            const int m = ti * 16 + j0 + 8 * hi;  // C layout: vgpr j -> row j + 8*hi
            float v = c[t][j0];
            if (bias) v += bias[n];
            if (relu) v = v > 0.0f ? v : 0.0f;
            const long long idx = (long long)m * N + n;
            if (cf) cf[idx] = v;
            if (cb) cb[idx] = (bf16)v;
        }
    }
}

// ---------------------------------------------------------------------------
// Prep / pointwise kernels
// ---------------------------------------------------------------------------
__global__ void k_f32_to_bf16(const float* __restrict__ in, bf16* __restrict__ out, int n)
{
    int i = blockIdx.x * blockDim.x + threadIdx.x;
    if (i < n) out[i] = (bf16)in[i];
}

// WcatT[j, f]: j = r*64+h -> sum_b comp[r,b]*basis[b,f,h];  j in [1024,1088) -> root[f,h]
__global__ void k_build_wcat(const float* __restrict__ basis, const float* __restrict__ comp,
                             const float* __restrict__ root, bf16* __restrict__ WcatT)
{
    int gid = blockIdx.x * blockDim.x + threadIdx.x;
    if (gid >= WCAT_N * N_FEAT) return;
    int j = gid >> 9, f = gid & (N_FEAT - 1);
    int r = j >> 6, h = j & 63;
    float v;
    if (r < N_REL) {
        v = 0.0f;
        for (int b = 0; b < N_BASES; ++b)
            v += comp[r * N_BASES + b] * basis[(b * N_FEAT + f) * HID + h];
    } else {
        v = root[f * HID + h];
    }
    WcatT[(long long)j * N_FEAT + f] = (bf16)v;
}

// wnrT[n,k] (n<64 -> w_nbr[k,n]; n>=64 -> w_root[k,n-64])
__global__ void k_build_wnr(const float* __restrict__ w_nbr, const float* __restrict__ w_root,
                            bf16* __restrict__ wnrT)
{
    int gid = blockIdx.x * blockDim.x + threadIdx.x;
    if (gid >= 128 * HID) return;
    int n = gid >> 6, k = gid & 63;
    float v = (n < 64) ? w_nbr[k * 64 + n] : w_root[k * 64 + (n - 64)];
    wnrT[gid] = (bf16)v;
}

// out[c*rows + r] = in[r*cols + c]  (f32 -> bf16 transpose)
__global__ void k_transpose_b(const float* __restrict__ in, bf16* __restrict__ out,
                              int rows, int cols)
{
    int gid = blockIdx.x * blockDim.x + threadIdx.x;
    if (gid >= rows * cols) return;
    int r = gid / cols, c = gid % cols;
    out[(long long)c * rows + r] = (bf16)in[(long long)r * cols + c];
}

__global__ void k_deg(const int* __restrict__ dst, float* __restrict__ deg, int E)
{
    int e = blockIdx.x * blockDim.x + threadIdx.x;
    if (e < E) atomicAdd(&deg[dst[e]], 1.0f);
}

// agg[dst, h] += xw[src, etype*64 + h]   (xw row stride WCAT_N)
__global__ void k_scatter_rgcn(const int* __restrict__ src, const int* __restrict__ dst,
                               const int* __restrict__ et, const float* __restrict__ xw,
                               float* __restrict__ agg, int E)
{
    int gid = blockIdx.x * blockDim.x + threadIdx.x;
    int e = gid >> 6, h = gid & 63;
    if (e >= E) return;
    float v = xw[(long long)src[e] * WCAT_N + et[e] * 64 + h];
    atomicAdd(&agg[(long long)dst[e] * 64 + h], v);
}

// nb[dst, h] += t2[src, h]   (t2 row stride 128; first 64 cols = h1@w_nbr)
__global__ void k_scatter_nbr(const int* __restrict__ src, const int* __restrict__ dst,
                              const float* __restrict__ t2, float* __restrict__ nb, int E)
{
    int gid = blockIdx.x * blockDim.x + threadIdx.x;
    int e = gid >> 6, h = gid & 63;
    if (e >= E) return;
    atomicAdd(&nb[(long long)dst[e] * 64 + h], t2[(long long)src[e] * 128 + h]);
}

// h1 = agg/max(deg,1) + xw[:,1024:1088] + bias1
__global__ void k_h1(const float* __restrict__ agg, const float* __restrict__ deg,
                     const float* __restrict__ xw, const float* __restrict__ bias1,
                     float* __restrict__ h1f, bf16* __restrict__ h1b)
{
    int gid = blockIdx.x * blockDim.x + threadIdx.x;
    if (gid >= N_NODES * HID) return;
    int n = gid >> 6, h = gid & 63;
    float d = deg[n]; d = d > 1.0f ? d : 1.0f;
    float v = agg[gid] / d + xw[(long long)n * WCAT_N + 1024 + h] + bias1[h];
    h1f[gid] = v;
    h1b[gid] = (bf16)v;
}

// emotions = [x | nb + t2[:,64:128] + bias2]; also bf16 copy and umask-scaled copy
__global__ void k_emotions(const float* __restrict__ x, const float* __restrict__ nb,
                           const float* __restrict__ t2, const float* __restrict__ bias2,
                           const float* __restrict__ umask,
                           float* __restrict__ emo, bf16* __restrict__ emob,
                           bf16* __restrict__ emoM)
{
    int gid = blockIdx.x * blockDim.x + threadIdx.x;
    if (gid >= N_NODES * D_EMO) return;
    int n = gid / D_EMO, d = gid % D_EMO;
    float v;
    if (d < N_FEAT) v = x[(long long)n * N_FEAT + d];
    else            v = nb[n * 64 + (d - 512)] + t2[n * 128 + 64 + (d - 512)] + bias2[d - 512];
    emo[gid]  = v;
    emob[gid] = (bf16)v;
    int b = n >> 7, t = n & 127;
    emoM[gid] = (bf16)(v * umask[b * SEQ_L + t]);
}

// emoT[b, d, k] = emotions[b*128 + k, d]   (per-conversation transpose, bf16)
__global__ void k_emoT(const float* __restrict__ emo, bf16* __restrict__ emoT)
{
    int gid = blockIdx.x * blockDim.x + threadIdx.x;
    if (gid >= B_CONV * D_EMO * SEQ_L) return;
    int b = gid / (D_EMO * SEQ_L);
    int rem = gid % (D_EMO * SEQ_L);
    int d = rem / SEQ_L, k = rem % SEQ_L;
    emoT[gid] = (bf16)emo[((long long)b * SEQ_L + k) * D_EMO + d];
}

// alpha = renorm( softmax(tanh(scoresRaw * umask_k)) * umask_k )
__global__ void k_softmax(const float* __restrict__ scores, const float* __restrict__ umask,
                          bf16* __restrict__ alpha)
{
    int row = blockIdx.x;                 // b*128 + q
    int b = row >> 7;
    int k = threadIdx.x;                  // 0..127
    __shared__ float sh[128];
    float um = umask[b * SEQ_L + k];
    float a = tanhf(scores[(long long)row * SEQ_L + k] * um);
    sh[k] = a; __syncthreads();
    for (int off = 64; off > 0; off >>= 1) { if (k < off) sh[k] = fmaxf(sh[k], sh[k + off]); __syncthreads(); }
    float mx = sh[0]; __syncthreads();
    float e = __expf(a - mx);
    sh[k] = e; __syncthreads();
    for (int off = 64; off > 0; off >>= 1) { if (k < off) sh[k] += sh[k + off]; __syncthreads(); }
    float se = sh[0]; __syncthreads();
    float al = (e / se) * um;
    sh[k] = al; __syncthreads();
    for (int off = 64; off > 0; off >>= 1) { if (k < off) sh[k] += sh[k + off]; __syncthreads(); }
    float sa = sh[0];
    alpha[(long long)row * SEQ_L + k] = (bf16)(al / sa);
}

// logits = hidden @ Ws + bs; out = log_softmax(logits)
__global__ void k_head(const float* __restrict__ hidden, const float* __restrict__ Ws,
                       const float* __restrict__ bs, float* __restrict__ out)
{
    int n = blockIdx.x * blockDim.x + threadIdx.x;
    if (n >= N_NODES) return;
    float lg[N_CLS];
    for (int c = 0; c < N_CLS; ++c) {
        float v = bs[c];
        for (int h = 0; h < HID; ++h) v += hidden[n * HID + h] * Ws[h * N_CLS + c];
        lg[c] = v;
    }
    float mx = lg[0];
    for (int c = 1; c < N_CLS; ++c) mx = fmaxf(mx, lg[c]);
    float s = 0.0f;
    for (int c = 0; c < N_CLS; ++c) s += __expf(lg[c] - mx);
    float ls = __logf(s);
    for (int c = 0; c < N_CLS; ++c) out[n * N_CLS + c] = lg[c] - mx - ls;
}

// ---------------------------------------------------------------------------
static inline void launch_gemm(const bf16* A, const bf16* Bt, float* Cf, bf16* Cb,
                               const float* bias, int M, int N, int K,
                               long long aS, long long bS, long long cS,
                               int batches, int relu, hipStream_t stream)
{
    int tiles = (M >> 4) * (N >> 6);   // 16x64 C tile per wave
    dim3 grid((tiles + 7) / 8, batches);
    gemm_bf16_wmma<<<grid, 256, 0, stream>>>(A, Bt, Cf, Cb, bias, M, N, K, aS, bS, cS, relu);
}

extern "C" void kernel_launch(void* const* d_in, const int* in_sizes, int n_in,
                              void* d_out, int out_size, void* d_ws, size_t ws_size,
                              hipStream_t stream)
{
    const float* x      = (const float*)d_in[0];
    const int*   eidx   = (const int*)d_in[1];
    const int*   src    = eidx;
    const int*   dst    = eidx + N_EDGE;
    const int*   etype  = (const int*)d_in[3];
    const float* umask  = (const float*)d_in[5];
    const float* basis  = (const float*)d_in[8];
    const float* comp   = (const float*)d_in[9];
    const float* root   = (const float*)d_in[10];
    const float* bias1  = (const float*)d_in[11];
    const float* w_nbr  = (const float*)d_in[12];
    const float* w_root = (const float*)d_in[13];
    const float* bias2  = (const float*)d_in[14];
    const float* Wm     = (const float*)d_in[15];
    const float* bm     = (const float*)d_in[16];
    const float* Wl     = (const float*)d_in[17];
    const float* bl     = (const float*)d_in[18];
    const float* Ws     = (const float*)d_in[19];
    const float* bs     = (const float*)d_in[20];
    float* out = (float*)d_out;

    // ---- workspace carve-up ----
    char* w = (char*)d_ws;
    auto carve = [&](size_t bytes) { char* p = w; w += (bytes + 255) & ~(size_t)255; return p; };
    bf16*  x_b    = (bf16*)carve((size_t)N_NODES * N_FEAT * 2);
    bf16*  WcatT  = (bf16*)carve((size_t)WCAT_N * N_FEAT * 2);
    float* xw     = (float*)carve((size_t)N_NODES * WCAT_N * 4);
    float* deg    = (float*)carve((size_t)N_NODES * 4);
    float* agg    = (float*)carve((size_t)N_NODES * HID * 4);
    float* h1f    = (float*)carve((size_t)N_NODES * HID * 4);
    bf16*  h1b    = (bf16*)carve((size_t)N_NODES * HID * 2);
    bf16*  wnrT   = (bf16*)carve((size_t)128 * HID * 2);
    float* t2     = (float*)carve((size_t)N_NODES * 128 * 4);
    float* nb     = (float*)carve((size_t)N_NODES * HID * 4);
    float* emo    = (float*)carve((size_t)N_NODES * D_EMO * 4);
    bf16*  emob   = (bf16*)carve((size_t)N_NODES * D_EMO * 2);
    bf16*  emoM   = (bf16*)carve((size_t)N_NODES * D_EMO * 2);
    bf16*  emoT   = (bf16*)carve((size_t)B_CONV * D_EMO * SEQ_L * 2);
    bf16*  WmT    = (bf16*)carve((size_t)D_EMO * D_EMO * 2);
    bf16*  Qb     = (bf16*)carve((size_t)N_NODES * D_EMO * 2);
    float* scores = (float*)carve((size_t)B_CONV * SEQ_L * SEQ_L * 4);
    bf16*  alphab = (bf16*)carve((size_t)B_CONV * SEQ_L * SEQ_L * 2);
    bf16*  attb   = (bf16*)carve((size_t)N_NODES * D_EMO * 2);
    bf16*  WlT    = (bf16*)carve((size_t)HID * D_EMO * 2);
    float* hidden = (float*)carve((size_t)N_NODES * HID * 4);

    // ---- zero the accumulators used by atomics ----
    hipMemsetAsync(deg, 0, (size_t)N_NODES * 4, stream);
    hipMemsetAsync(agg, 0, (size_t)N_NODES * HID * 4, stream);
    hipMemsetAsync(nb,  0, (size_t)N_NODES * HID * 4, stream);

    const int T = 256;
    auto cdiv = [](long long a, long long b) { return (int)((a + b - 1) / b); };

    // ---- RGCN: fused basis+root GEMM ----
    k_f32_to_bf16<<<cdiv((long long)N_NODES * N_FEAT, T), T, 0, stream>>>(x, x_b, N_NODES * N_FEAT);
    k_build_wcat<<<cdiv((long long)WCAT_N * N_FEAT, T), T, 0, stream>>>(basis, comp, root, WcatT);
    launch_gemm(x_b, WcatT, xw, nullptr, nullptr,
                N_NODES, WCAT_N, N_FEAT, 0, 0, 0, 1, 0, stream);

    // ---- edge aggregation (mean over typed messages) ----
    k_deg<<<cdiv(N_EDGE, T), T, 0, stream>>>(dst, deg, N_EDGE);
    k_scatter_rgcn<<<cdiv((long long)N_EDGE * 64, T), T, 0, stream>>>(src, dst, etype, xw, agg, N_EDGE);
    k_h1<<<cdiv((long long)N_NODES * HID, T), T, 0, stream>>>(agg, deg, xw, bias1, h1f, h1b);

    // ---- GraphConv: fused [w_nbr | w_root] GEMM + scatter ----
    k_build_wnr<<<cdiv(128 * HID, T), T, 0, stream>>>(w_nbr, w_root, wnrT);
    launch_gemm(h1b, wnrT, t2, nullptr, nullptr,
                N_NODES, 128, HID, 0, 0, 0, 1, 0, stream);
    k_scatter_nbr<<<cdiv((long long)N_EDGE * 64, T), T, 0, stream>>>(src, dst, t2, nb, N_EDGE);

    // ---- emotions + layouts for attention ----
    k_emotions<<<cdiv((long long)N_NODES * D_EMO, T), T, 0, stream>>>(x, nb, t2, bias2, umask,
                                                                     emo, emob, emoM);
    k_emoT<<<cdiv((long long)B_CONV * D_EMO * SEQ_L, T), T, 0, stream>>>(emo, emoT);

    // ---- Q = emotions @ Wm + bm (bf16 out) ----
    k_transpose_b<<<cdiv(D_EMO * D_EMO, T), T, 0, stream>>>(Wm, WmT, D_EMO, D_EMO);
    launch_gemm(emob, WmT, nullptr, Qb, bm,
                N_NODES, D_EMO, D_EMO, 0, 0, 0, 1, 0, stream);

    // ---- batched scores = Q @ Mmask^T per conversation ----
    launch_gemm(Qb, emoM, scores, nullptr, nullptr,
                SEQ_L, SEQ_L, D_EMO,
                (long long)SEQ_L * D_EMO, (long long)SEQ_L * D_EMO,
                (long long)SEQ_L * SEQ_L, B_CONV, 0, stream);

    // ---- masked tanh-softmax with renormalization ----
    k_softmax<<<N_NODES, SEQ_L, 0, stream>>>(scores, umask, alphab);

    // ---- batched att = alpha @ M per conversation (bf16 out) ----
    launch_gemm(alphab, emoT, nullptr, attb, nullptr,
                SEQ_L, D_EMO, SEQ_L,
                (long long)SEQ_L * SEQ_L, (long long)D_EMO * SEQ_L,
                (long long)SEQ_L * D_EMO, B_CONV, 0, stream);

    // ---- hidden = relu(att @ Wl + bl) ----
    k_transpose_b<<<cdiv(D_EMO * HID, T), T, 0, stream>>>(Wl, WlT, D_EMO, HID);
    launch_gemm(attb, WlT, hidden, nullptr, bl,
                N_NODES, HID, D_EMO, 0, 0, 0, 1, 1, stream);

    // ---- classifier + log_softmax ----
    k_head<<<cdiv(N_NODES, T), T, 0, stream>>>(hidden, Ws, bs, out);
}
